// ParseridgeModel_15101105013315
// MI455X (gfx1250) — matrix operand order repaired and checked by reading, not compiled, not run on hardware
//
#include <hip/hip_runtime.h>

typedef _Float16 half2t  __attribute__((ext_vector_type(2)));
typedef _Float16 half8   __attribute__((ext_vector_type(8)));
typedef _Float16 half16  __attribute__((ext_vector_type(16)));
typedef float    f32x2   __attribute__((ext_vector_type(2)));
typedef float    f32x8   __attribute__((ext_vector_type(8)));

#define B_TOT    16384
#define SEQ      64
#define DIM      250       // per-slot feature dim
#define KTOT     1000      // 4 * DIM
#define KPAD     1024
#define KCH      128       // K chunk staged in LDS
#define ROWS     64        // parser states per block
#define N_TRANS  4
#define N_LAB    50
#define N_VALID  54
#define NPAD     64        // 54 cols padded to 4 WMMA N-tiles
#define LSTRIDE  (KCH + 8) // LDS row stride in halves (bank-conflict pad)
#define NTHREADS 128       // 4 wave32s
#define NCHUNK   (KPAD / KCH)                      // 8
#define PAIRS_T  ((ROWS * (KCH / 2)) / NTHREADS)   // 32 float2 per thread per chunk
#define KSTEPS   (KPAD / 32)                       // 32 WMMA k-steps total
#define WP_HALVES (KSTEPS * 4 * 32 * 16)           // 65536 halves = 128 KB

// ---------------------------------------------------------------------------
// One-time per launch: pack [W_t ; W_r] (f32) into f16 in exact WMMA
// B-fragment order Wp[ks][nt][lane][j], and pack biases to 64 floats.
// Consumer: lane needs halves j=0..15 of W[n = nt*16 + (lane&15)]
//           [k = ks*32 + 16*(j>=8) + 8*(lane>=16) + (j&7)]  -> one contiguous 32B run/lane.
// ---------------------------------------------------------------------------
__global__ __launch_bounds__(256) void pack_weights(
    const float* __restrict__ W_t, const float* __restrict__ b_t,
    const float* __restrict__ W_r, const float* __restrict__ b_r,
    _Float16* __restrict__ Wp, float* __restrict__ biasPk)
{
    int g = blockIdx.x * 256 + threadIdx.x;      // 0 .. WP_HALVES-1
    int j    = g & 15;
    int lane = (g >> 4) & 31;
    int nt   = (g >> 9) & 3;
    int ks   = g >> 11;
    int n = nt * 16 + (lane & 15);
    int k = ks * 32 + ((j >> 3) << 4) + ((lane >> 4) << 3) + (j & 7);
    float v = 0.0f;
    if (n < N_VALID && k < KTOT)
        v = (n < N_TRANS) ? W_t[n * KTOT + k] : W_r[(n - N_TRANS) * KTOT + k];
    Wp[g] = (_Float16)v;

    if (g < NPAD) {
        float bv = 0.0f;
        if (g < N_TRANS)        bv = b_t[g];
        else if (g < N_VALID)   bv = b_r[g - N_TRANS];
        biasPk[g] = bv;
    }
}

// ---------------------------------------------------------------------------
// Pipeline helpers: register-prefetch a gather chunk, then convert+store it.
// ---------------------------------------------------------------------------
__device__ __forceinline__ void load_chunk(
    int c, int t, const float* __restrict__ lstm_out,
    const int (*srcOff)[4], f32x2* vals)
{
    #pragma unroll
    for (int i = 0; i < PAIRS_T; ++i) {
        int p   = t + i * NTHREADS;
        int row = p >> 6;            // 64 pairs per row
        int pi  = p & 63;
        int k   = c * KCH + pi * 2;
        f32x2 v = {};
        if (k < KTOT) {
            int seg = (k >= 750) ? 3 : (k >= 500) ? 2 : (k >= 250) ? 1 : 0;
            int off = k - seg * DIM;
            v = *(const f32x2*)(lstm_out + srcOff[row][seg] + off);
        }
        vals[i] = v;
    }
}

__device__ __forceinline__ void store_chunk(
    int c, int t, const float (*srcScale)[4],
    const f32x2* vals, _Float16 (*buf)[LSTRIDE])
{
    #pragma unroll
    for (int i = 0; i < PAIRS_T; ++i) {
        int p   = t + i * NTHREADS;
        int row = p >> 6;
        int pi  = p & 63;
        int k   = c * KCH + pi * 2;
        float sc = 0.0f;
        if (k < KTOT) {
            int seg = (k >= 750) ? 3 : (k >= 500) ? 2 : (k >= 250) ? 1 : 0;
            sc = srcScale[row][seg];
        }
        half2t h;
        h.x = (_Float16)(vals[i].x * sc);
        h.y = (_Float16)(vals[i].y * sc);
        *(half2t*)&buf[row][pi * 2] = h;
    }
}

// ---------------------------------------------------------------------------
// Main kernel: gather -> LDS (double-buffered) -> WMMA, B frags from packed
// global (L2-resident), tanh epilogue.
// ---------------------------------------------------------------------------
__global__ __launch_bounds__(NTHREADS) void parser_mlp_wmma(
    const float* __restrict__ lstm_out,
    const _Float16* __restrict__ Wp, const float* __restrict__ biasPk,
    const int*   __restrict__ stack_index, const int* __restrict__ buffer_index,
    const int*   __restrict__ stack_len,   const int* __restrict__ buffer_len,
    float* __restrict__ out_trans, float* __restrict__ out_rel)
{
    __shared__ _Float16 As[2][ROWS][LSTRIDE];  // double-buffered activations
    __shared__ int      srcOff[ROWS][4];
    __shared__ float    srcScale[ROWS][4];

    const int t      = threadIdx.x;
    const int block0 = blockIdx.x * ROWS;
    const int wave   = t >> 5;
    const int lane   = t & 31;
    const int lhalf  = lane >> 4;
    const int l16    = lane & 15;

    // per-row gather descriptors (3 stack slots + 1 buffer slot)
    #pragma unroll
    for (int j = 0; j < (ROWS * 4) / NTHREADS; ++j) {
        int tt = t + j * NTHREADS;
        int r = tt >> 2, s = tt & 3;
        int b = block0 + r;
        int idx, valid;
        if (s < 3) { idx = stack_index[b * 3 + s]; valid = (s < stack_len[b]); }
        else       { idx = buffer_index[b];        valid = (buffer_len[b] > 0); }
        srcOff[r][s]   = (b * SEQ + idx) * DIM;
        srcScale[r][s] = valid ? 1.0f : 0.0f;
    }
    __syncthreads();   // descriptors visible before first gather

    f32x2 vals[PAIRS_T];
    f32x8 acc[4] = {};

    // prologue: stage chunk 0
    load_chunk(0, t, lstm_out, srcOff, vals);
    store_chunk(0, t, srcScale, vals, As[0]);
    __syncthreads();

    const int mrow = wave * 16 + l16;
    for (int c = 0; c < NCHUNK; ++c) {
        // issue next chunk's gather loads first (hidden under WMMA below)
        if (c + 1 < NCHUNK)
            load_chunk(c + 1, t, lstm_out, srcOff, vals);

        const _Float16 (*__restrict__ buf)[LSTRIDE] = As[c & 1];
        #pragma unroll
        for (int s = 0; s < KCH / 32; ++s) {
            const int kc = s * 32;
            const int ks = c * (KCH / 32) + s;
            // A frag (16x32 f16): two 16B LDS runs per ISA layout
            half8 alo = *(const half8*)&buf[mrow][kc + lhalf * 8];
            half8 ahi = *(const half8*)&buf[mrow][kc + lhalf * 8 + 16];
            half16 a = __builtin_shufflevector(alo, ahi,
                        0,1,2,3,4,5,6,7,8,9,10,11,12,13,14,15);
            #pragma unroll
            for (int nt = 0; nt < 4; ++nt) {
                // B frag: pre-packed, 32B contiguous per lane, L2-resident
                const half16 bb = *(const half16*)(Wp +
                    ((((ks * 4 + nt) * 32) + lane) << 4));
                acc[nt] = __builtin_amdgcn_wmma_f32_16x16x32_f16(
                    false, a, false, bb, (short)0, acc[nt], false, false);
            }
        }

        if (c + 1 < NCHUNK)
            store_chunk(c + 1, t, srcScale, vals, As[(c + 1) & 1]);
        __syncthreads();
    }

    // epilogue: bias + tanh, scatter to concatenated outputs
    #pragma unroll
    for (int nt = 0; nt < 4; ++nt) {
        int n = nt * 16 + l16;
        float bias = biasPk[n];
        #pragma unroll
        for (int r = 0; r < 8; ++r) {
            int m = block0 + wave * 16 + r + lhalf * 8;  // C/D: M = r + 8*hi
            float v = tanhf(acc[nt][r] + bias);
            if (n < N_TRANS)      out_trans[m * N_TRANS + n] = v;
            else if (n < N_VALID) out_rel[m * N_LAB + (n - N_TRANS)] = v;
        }
    }
}

extern "C" void kernel_launch(void* const* d_in, const int* in_sizes, int n_in,
                              void* d_out, int out_size, void* d_ws, size_t ws_size,
                              hipStream_t stream) {
    const float* lstm_out     = (const float*)d_in[0];
    const float* W_t          = (const float*)d_in[1];
    const float* b_t          = (const float*)d_in[2];
    const float* W_r          = (const float*)d_in[3];
    const float* b_r          = (const float*)d_in[4];
    const int*   stack_index  = (const int*)d_in[5];
    const int*   buffer_index = (const int*)d_in[6];
    const int*   stack_len    = (const int*)d_in[7];
    const int*   buffer_len   = (const int*)d_in[8];

    _Float16* Wp     = (_Float16*)d_ws;                          // 128 KB
    float*    biasPk = (float*)((char*)d_ws + WP_HALVES * 2);    // +256 B

    pack_weights<<<dim3(WP_HALVES / 256), dim3(256), 0, stream>>>(
        W_t, b_t, W_r, b_r, Wp, biasPk);

    float* out_trans = (float*)d_out;                        // [B, 4]
    float* out_rel   = out_trans + (size_t)B_TOT * N_TRANS;  // [B, 50]

    parser_mlp_wmma<<<dim3(B_TOT / ROWS), dim3(NTHREADS), 0, stream>>>(
        lstm_out, Wp, biasPk,
        stack_index, buffer_index, stack_len, buffer_len,
        out_trans, out_rel);
}
